// EncoderLayer_67010079752440
// MI455X (gfx1250) — matrix operand order repaired and checked by reading, not compiled, not run on hardware
//
#include <hip/hip_runtime.h>
#include <hip/hip_bf16.h>
#include <cstdint>

// MI455X (gfx1250) transformer encoder layer.
// - All matmuls: v_wmma_f32_16x16x32_bf16 (fp32 accum), compute-bound layer
//   (~276 GFLOP vs ~0.5 GB HBM traffic @ 23.3 TB/s => WMMA is the ceiling).
// - GEMM staging uses CDNA5 async global->LDS DMA (ASYNCcnt) with
//   double-buffered LDS tiles: stage(k+1) overlaps WMMA(k).
// - Attention is flash-style (online softmax): the [4,16,2048,2048] score
//   tensor never touches HBM.

typedef __attribute__((ext_vector_type(16))) __bf16 v16bf;
typedef __attribute__((ext_vector_type(8)))  float  v8f;

namespace {
constexpr int kB = 4, kS = 2048, kH = 1024, kNH = 16, kHD = 64, kPF = 4096;
constexpr float kEPS = 1e-5f;
}

// ---------------------------------------------------------------------------
// fp32 -> bf16 convert (straight copy)
// ---------------------------------------------------------------------------
__global__ void cvt_f32_to_bf16(const float* __restrict__ in,
                                __bf16* __restrict__ out, long n) {
  long i = (long)blockIdx.x * blockDim.x + threadIdx.x;
  if (i < n) out[i] = (__bf16)in[i];
}

// ---------------------------------------------------------------------------
// fp32 [K][N] -> bf16 [N][K] (weights pre-transposed so the WMMA B-fragment
// layout -- lane=column n, k contiguous -- streams from contiguous memory)
// ---------------------------------------------------------------------------
__global__ __launch_bounds__(256)
void transpose_cvt(const float* __restrict__ in, __bf16* __restrict__ out,
                   int K, int N) {
  __shared__ float tile[32][33];
  const int tx = threadIdx.x & 31;
  const int ty = threadIdx.x >> 5;   // 0..7
  const int k0 = blockIdx.x * 32, n0 = blockIdx.y * 32;
#pragma unroll
  for (int i = 0; i < 32; i += 8) {
    int k = k0 + ty + i, n = n0 + tx;
    tile[ty + i][tx] = (k < K && n < N) ? in[(long)k * N + n] : 0.0f;
  }
  __syncthreads();
#pragma unroll
  for (int i = 0; i < 32; i += 8) {
    int n = n0 + ty + i, k = k0 + tx;
    if (n < N && k < K) out[(long)n * K + k] = (__bf16)tile[tx][ty + i];
  }
}

// ---------------------------------------------------------------------------
// V [B,S,NH*HD] -> Vt [B,NH,HD,S]  (so attn@V B-fragments are contiguous)
// ---------------------------------------------------------------------------
__global__ void v_transpose(const __bf16* __restrict__ V,
                            __bf16* __restrict__ Vt, long total) {
  long i = (long)blockIdx.x * blockDim.x + threadIdx.x;
  if (i >= total) return;
  int s = (int)(i % kS);
  long t = i / kS;
  int d = (int)(t % kHD); t /= kHD;
  int h = (int)(t % kNH);
  int b = (int)(t / kNH);
  Vt[i] = V[((long)(b * kS + s)) * kH + h * kHD + d];
}

// ---------------------------------------------------------------------------
// Generic bf16 WMMA GEMM:  C[z] = act(A[z] (MxK) * Bt[z]^T (stored NxK) + bias)
//                                 (+ resid)   -> outF (f32) and/or outB (bf16)
// Block tile 128x128x32, 8 waves (2x4), each wave -> 4x2 16x16 accum tiles.
// Global->LDS staging via CDNA5 async DMA (ASYNCcnt), double-buffered.
// ---------------------------------------------------------------------------
__global__ __launch_bounds__(256)
void gemm_bf16_wmma(const __bf16* __restrict__ A, int lda, long aZ,
                    const __bf16* __restrict__ Bt, long bZ,
                    const float* __restrict__ bias,
                    const float* __restrict__ resid, long rZ,
                    float* __restrict__ outF, __bf16* __restrict__ outB,
                    int ldc, long cZ, int M, int N, int K, int doRelu) {
  __shared__ __bf16 As[2][128 * 40];   // row stride 40 bf16 = 80B (16B align)
  __shared__ __bf16 Bs[2][128 * 40];   // stored [n][k]

  const int tid = threadIdx.x;
  const int lane = tid & 31, wid = tid >> 5;
  const int wr = wid >> 2, wc = wid & 3;       // 2x4 wave grid
  const int hl = lane >> 4, l15 = lane & 15;
  const long m0 = (long)blockIdx.x * 128;
  const int n0 = blockIdx.y * 128;
  const long z = blockIdx.z;

  const __bf16* Az = A + aZ * z;
  const __bf16* Bz = Bt + bZ * z;

  const int ar = tid >> 1;          // 0..127: tile row handled by this thread
  const int ac = (tid & 1) * 16;    // 0 or 16: half-row (16 bf16 = 32B)
  // Clamp out-of-range B rows to N-1 so every stage issues exactly 4 async
  // ops per thread (uniform ASYNCcnt accounting). Duplicate rows only feed
  // WMMA columns that the epilogue discards (n >= N).
  const int br = (n0 + ar < N) ? (n0 + ar) : (N - 1);

  // Issue one 128x32 A tile + 128x32 B tile (bf16) for k-step `kt` into
  // buffer `buf` using async global->LDS DMA: 4 x b128 per thread.
  auto stage = [&](int buf, int kt) {
    unsigned la = (unsigned)(uintptr_t)(&As[buf][ar * 40 + ac]);
    unsigned lb = (unsigned)(uintptr_t)(&Bs[buf][ar * 40 + ac]);
    unsigned long long ga =
        (unsigned long long)(uintptr_t)(Az + (m0 + ar) * (long)lda + kt + ac);
    unsigned long long gb =
        (unsigned long long)(uintptr_t)(Bz + (long)br * K + kt + ac);
    asm volatile(
        "global_load_async_to_lds_b128 %0, %2, off\n\t"
        "global_load_async_to_lds_b128 %1, %3, off\n\t"
        "global_load_async_to_lds_b128 %4, %6, off\n\t"
        "global_load_async_to_lds_b128 %5, %7, off"
        :
        : "v"(la), "v"(la + 16u), "v"(ga), "v"(ga + 16ull),
          "v"(lb), "v"(lb + 16u), "v"(gb), "v"(gb + 16ull)
        : "memory");
  };

  v8f acc[4][2];
#pragma unroll
  for (int i = 0; i < 4; ++i)
#pragma unroll
    for (int j = 0; j < 2; ++j) acc[i][j] = (v8f)(0.0f);

  stage(0, 0);
  if (32 < K) stage(1, 32);

  for (int kt = 0; kt < K; kt += 32) {
    const int buf = (kt >> 5) & 1;
    // Retire this buffer's stage: async completion is in-order; at most one
    // later stage (4 ops) may remain outstanding.
    if (kt + 32 < K) {
      asm volatile("s_wait_asynccnt 0x4" ::: "memory");
    } else {
      asm volatile("s_wait_asynccnt 0x0" ::: "memory");
    }
    __syncthreads();

    const __bf16* Ab = As[buf];
    const __bf16* Bb = Bs[buf];

    // A fragment: lane m=l15; k = 16*hl + e (contiguous 16)
    v16bf afr[4];
#pragma unroll
    for (int mt = 0; mt < 4; ++mt) {
      const __bf16* p = Ab + (wr * 64 + mt * 16 + l15) * 40;
      ((uint4*)&afr[mt])[0] = *(const uint4*)(p + 16 * hl);
      ((uint4*)&afr[mt])[1] = *(const uint4*)(p + 16 * hl + 8);
    }
    // B fragment: lane n=l15; k = 16*hl + e (contiguous 16)
    v16bf bfr[2];
#pragma unroll
    for (int nt = 0; nt < 2; ++nt) {
      const __bf16* p = Bb + (wc * 32 + nt * 16 + l15) * 40 + 16 * hl;
      ((uint4*)&bfr[nt])[0] = *(const uint4*)(p);
      ((uint4*)&bfr[nt])[1] = *(const uint4*)(p + 8);
    }
#pragma unroll
    for (int mt = 0; mt < 4; ++mt)
#pragma unroll
      for (int nt = 0; nt < 2; ++nt)
        acc[mt][nt] = __builtin_amdgcn_wmma_f32_16x16x32_bf16(
            false, afr[mt], false, bfr[nt], (short)0, acc[mt][nt], false, false);

    __syncthreads();                 // all waves done reading `buf`
    if (kt + 64 < K) stage(buf, kt + 64);
  }

  // Epilogue: C layout is VGPR r <-> row r + 8*hl, col = lane&15.
#pragma unroll
  for (int mt = 0; mt < 4; ++mt)
#pragma unroll
    for (int nt = 0; nt < 2; ++nt) {
      int n = n0 + wc * 32 + nt * 16 + l15;
      if (n >= N) continue;
      float bi = bias ? bias[n] : 0.0f;
#pragma unroll
      for (int r = 0; r < 8; ++r) {
        long m = m0 + wr * 64 + mt * 16 + r + 8 * hl;
        float v = acc[mt][nt][r] + bi;
        if (doRelu) v = fmaxf(v, 0.0f);
        long idx = cZ * z + m * (long)ldc + n;
        if (resid) v += resid[rZ * z + m * (long)ldc + n];
        if (outF) outF[idx] = v;
        if (outB) outB[idx] = (__bf16)v;
      }
    }
}

// ---------------------------------------------------------------------------
// Flash attention: softmax(Q Kp^T) V, per (b, head, 128-q-row block).
// Each wave owns 16 q rows; loop over keys in steps of 32.
// Q/Kp: bf16 [B,S,H]; Vt: bf16 [B,NH,HD,S]; O: bf16 [B,S,H].
// ---------------------------------------------------------------------------
__global__ __launch_bounds__(256)
void flash_attn_wmma(const __bf16* __restrict__ Q, const __bf16* __restrict__ Kp,
                     const __bf16* __restrict__ Vt,
                     const unsigned char* __restrict__ mask,
                     __bf16* __restrict__ O) {
  __shared__ __bf16 Ps[8 * 16 * 32];  // 1KB per-wave scratch for P relayout
  const int tid = threadIdx.x;
  const int lane = tid & 31, wid = tid >> 5;
  const int hl = lane >> 4, l15 = lane & 15;
  const int b = blockIdx.z, h = blockIdx.y;
  const int qbase = blockIdx.x * 128 + wid * 16;

  // Q A-fragments for k-steps c=0,1 over HD=64
  v16bf aq[2];
  {
    const __bf16* qp = Q + ((long)(b * kS + qbase + l15)) * kH + h * kHD;
#pragma unroll
    for (int c = 0; c < 2; ++c) {
      ((uint4*)&aq[c])[0] = *(const uint4*)(qp + 32 * c + 16 * hl);
      ((uint4*)&aq[c])[1] = *(const uint4*)(qp + 32 * c + 16 * hl + 8);
    }
  }

  v8f o[4];
#pragma unroll
  for (int i = 0; i < 4; ++i) o[i] = (v8f)(0.0f);
  float mrun[8], lrun[8];
#pragma unroll
  for (int r = 0; r < 8; ++r) { mrun[r] = -1e30f; lrun[r] = 0.0f; }

  __bf16* ps = Ps + wid * (16 * 32);

  for (int j = 0; j < kS; j += 32) {
    // scores S = Q (16x64) @ Kp_j^T (64x32): two 16x16 C tiles
    v8f s[2];
#pragma unroll
    for (int nt = 0; nt < 2; ++nt) {
      s[nt] = (v8f)(0.0f);
      const __bf16* kp =
          Kp + ((long)(b * kS + j + nt * 16 + l15)) * kH + h * kHD;
#pragma unroll
      for (int c = 0; c < 2; ++c) {
        v16bf bk;
        ((uint4*)&bk)[0] = *(const uint4*)(kp + 32 * c + 16 * hl);
        ((uint4*)&bk)[1] = *(const uint4*)(kp + 32 * c + 16 * hl + 8);
        s[nt] = __builtin_amdgcn_wmma_f32_16x16x32_bf16(
            false, aq[c], false, bk, (short)0, s[nt], false, false);
      }
      // mask: each lane holds one key column for all its rows
      if (!mask[b * kS + j + nt * 16 + l15]) {
#pragma unroll
        for (int r = 0; r < 8; ++r) s[nt][r] = -1e10f;
      }
    }
    // online softmax: row max over the 32 keys (cross 16-lane group + 2 tiles)
    float scale[8];
#pragma unroll
    for (int r = 0; r < 8; ++r) {
      float mx = fmaxf(s[0][r], s[1][r]);
#pragma unroll
      for (int off = 1; off < 16; off <<= 1)
        mx = fmaxf(mx, __shfl_xor(mx, off, 32));
      float mnew = fmaxf(mrun[r], mx);
      scale[r] = __expf(mrun[r] - mnew);
      mrun[r] = mnew;
    }
#pragma unroll
    for (int nt = 0; nt < 2; ++nt)
#pragma unroll
      for (int r = 0; r < 8; ++r) s[nt][r] = __expf(s[nt][r] - mrun[r]);
#pragma unroll
    for (int r = 0; r < 8; ++r) {
      float sm = s[0][r] + s[1][r];
#pragma unroll
      for (int off = 1; off < 16; off <<= 1) sm += __shfl_xor(sm, off, 32);
      lrun[r] = lrun[r] * scale[r] + sm;
    }
#pragma unroll
    for (int i = 0; i < 4; ++i)
#pragma unroll
      for (int r = 0; r < 8; ++r) o[i][r] *= scale[r];

    // P (C layout, fp32) -> bf16 A-fragment via per-wave LDS scratch.
    // Same-wave DS ops are in-order (ISA 7.3); compiler inserts dscnt waits.
#pragma unroll
    for (int nt = 0; nt < 2; ++nt)
#pragma unroll
      for (int r = 0; r < 8; ++r)
        ps[(r + 8 * hl) * 32 + nt * 16 + l15] = (__bf16)s[nt][r];
    __builtin_amdgcn_wave_barrier();
    v16bf pa;
    ((uint4*)&pa)[0] = *(const uint4*)(ps + l15 * 32 + 16 * hl);
    ((uint4*)&pa)[1] = *(const uint4*)(ps + l15 * 32 + 16 * hl + 8);

    // O += P (16x32) @ V_j (32x64): B-fragments stream from Vt contiguously
#pragma unroll
    for (int i = 0; i < 4; ++i) {
      const __bf16* vp =
          Vt + ((long)((b * kNH + h) * kHD + i * 16 + l15)) * kS + j + 16 * hl;
      v16bf bv;
      ((uint4*)&bv)[0] = *(const uint4*)(vp);
      ((uint4*)&bv)[1] = *(const uint4*)(vp + 8);
      o[i] = __builtin_amdgcn_wmma_f32_16x16x32_bf16(
          false, pa, false, bv, (short)0, o[i], false, false);
    }
  }

  float inv[8];
#pragma unroll
  for (int r = 0; r < 8; ++r) inv[r] = lrun[r] > 0.0f ? 1.0f / lrun[r] : 0.0f;
#pragma unroll
  for (int i = 0; i < 4; ++i)
#pragma unroll
    for (int r = 0; r < 8; ++r) {
      long row = qbase + r + 8 * hl;
      O[((long)(b * kS) + row) * kH + h * kHD + i * 16 + l15] =
          (__bf16)(o[i][r] * inv[r]);
    }
}

// ---------------------------------------------------------------------------
// LayerNorm over H=1024; writes fp32 result and optional bf16 copy.
// ---------------------------------------------------------------------------
__global__ __launch_bounds__(256)
void layernorm_k(const float* __restrict__ x, const float* __restrict__ g,
                 const float* __restrict__ be, float* __restrict__ outF,
                 __bf16* __restrict__ outB) {
  __shared__ float red[256];
  const long row = blockIdx.x;
  const float* xr = x + row * kH;
  float s = 0.0f;
  for (int i = threadIdx.x; i < kH; i += 256) s += xr[i];
  red[threadIdx.x] = s;
  __syncthreads();
  for (int st = 128; st > 0; st >>= 1) {
    if (threadIdx.x < st) red[threadIdx.x] += red[threadIdx.x + st];
    __syncthreads();
  }
  const float mu = red[0] * (1.0f / kH);
  __syncthreads();
  float v = 0.0f;
  for (int i = threadIdx.x; i < kH; i += 256) {
    float d = xr[i] - mu;
    v += d * d;
  }
  red[threadIdx.x] = v;
  __syncthreads();
  for (int st = 128; st > 0; st >>= 1) {
    if (threadIdx.x < st) red[threadIdx.x] += red[threadIdx.x + st];
    __syncthreads();
  }
  const float rs = rsqrtf(red[0] * (1.0f / kH) + kEPS);
  for (int i = threadIdx.x; i < kH; i += 256) {
    float y = (xr[i] - mu) * rs * g[i] + be[i];
    outF[row * kH + i] = y;
    if (outB) outB[row * kH + i] = (__bf16)y;
  }
}

// ---------------------------------------------------------------------------
extern "C" void kernel_launch(void* const* d_in, const int* in_sizes, int n_in,
                              void* d_out, int out_size, void* d_ws,
                              size_t ws_size, hipStream_t stream) {
  (void)in_sizes; (void)n_in; (void)out_size; (void)ws_size;
  const float* src = (const float*)d_in[0];
  const unsigned char* mask = (const unsigned char*)d_in[1];
  const float* Wq = (const float*)d_in[2];  const float* bq = (const float*)d_in[3];
  const float* Wk = (const float*)d_in[4];  const float* bk = (const float*)d_in[5];
  const float* Wv = (const float*)d_in[6];  const float* bv = (const float*)d_in[7];
  const float* Wmal = (const float*)d_in[8]; const float* bmal = (const float*)d_in[9];
  const float* Wo = (const float*)d_in[10]; const float* bo = (const float*)d_in[11];
  const float* W1 = (const float*)d_in[12]; const float* b1 = (const float*)d_in[13];
  const float* W2 = (const float*)d_in[14]; const float* b2 = (const float*)d_in[15];
  const float* g1 = (const float*)d_in[16]; const float* be1 = (const float*)d_in[17];
  const float* g2 = (const float*)d_in[18]; const float* be2 = (const float*)d_in[19];
  float* out = (float*)d_out;

  char* ws = (char*)d_ws;
  size_t off = 0;
  auto take = [&](size_t bytes) {
    void* p = ws + off;
    off = (off + bytes + 255) & ~(size_t)255;
    return p;
  };

  const long MSH = (long)kB * kS * kH;  // 8,388,608 tokens*feat
  __bf16* srcb = (__bf16*)take(MSH * 2);
  __bf16* wqt = (__bf16*)take((size_t)kH * kH * 2);
  __bf16* wkt = (__bf16*)take((size_t)kH * kH * 2);
  __bf16* wvt = (__bf16*)take((size_t)kH * kH * 2);
  __bf16* wot = (__bf16*)take((size_t)kH * kH * 2);
  __bf16* w1t = (__bf16*)take((size_t)kH * kPF * 2);
  __bf16* w2t = (__bf16*)take((size_t)kPF * kH * 2);
  __bf16* wmt = (__bf16*)take((size_t)kHD * kHD * 2);
  __bf16* qb = (__bf16*)take(MSH * 2);
  __bf16* kb = (__bf16*)take(MSH * 2);
  __bf16* kpb = (__bf16*)take(MSH * 2);
  __bf16* vb = (__bf16*)take(MSH * 2);
  __bf16* vtb = (__bf16*)take(MSH * 2);
  __bf16* attnb = (__bf16*)take(MSH * 2);
  float* y1 = (float*)take(MSH * 4);
  float* x1n = (float*)take(MSH * 4);
  __bf16* x1nb = (__bf16*)take(MSH * 2);
  __bf16* hb = (__bf16*)take((size_t)kB * kS * kPF * 2);
  float* y2 = (float*)take(MSH * 4);

  const int M = kB * kS;  // 8192

  // --- stage inputs to bf16 (weights transposed to [N][K]) ---
  cvt_f32_to_bf16<<<(MSH + 255) / 256, 256, 0, stream>>>(src, srcb, MSH);
  transpose_cvt<<<dim3(32, 32), 256, 0, stream>>>(Wq, wqt, kH, kH);
  transpose_cvt<<<dim3(32, 32), 256, 0, stream>>>(Wk, wkt, kH, kH);
  transpose_cvt<<<dim3(32, 32), 256, 0, stream>>>(Wv, wvt, kH, kH);
  transpose_cvt<<<dim3(32, 32), 256, 0, stream>>>(Wo, wot, kH, kH);
  transpose_cvt<<<dim3(32, 128), 256, 0, stream>>>(W1, w1t, kH, kPF);
  transpose_cvt<<<dim3(128, 32), 256, 0, stream>>>(W2, w2t, kPF, kH);
  transpose_cvt<<<dim3(2, 2), 256, 0, stream>>>(Wmal, wmt, kHD, kHD);

  // --- Q, K, V projections ---
  gemm_bf16_wmma<<<dim3(M / 128, kH / 128, 1), 256, 0, stream>>>(
      srcb, kH, 0, wqt, 0, bq, nullptr, 0, nullptr, qb, kH, 0, M, kH, kH, 0);
  gemm_bf16_wmma<<<dim3(M / 128, kH / 128, 1), 256, 0, stream>>>(
      srcb, kH, 0, wkt, 0, bk, nullptr, 0, nullptr, kb, kH, 0, M, kH, kH, 0);
  gemm_bf16_wmma<<<dim3(M / 128, kH / 128, 1), 256, 0, stream>>>(
      srcb, kH, 0, wvt, 0, bv, nullptr, 0, nullptr, vb, kH, 0, M, kH, kH, 0);

  // --- Kp = K @ Wmal + bmal, batched over 16 heads (z-stride 64 cols) ---
  gemm_bf16_wmma<<<dim3(M / 128, 1, kNH), 256, 0, stream>>>(
      kb, kH, kHD, wmt, 0, bmal, nullptr, 0, nullptr, kpb, kH, kHD,
      M, kHD, kHD, 0);

  // --- V -> [B,NH,HD,S] for contiguous attn@V B-fragments ---
  v_transpose<<<(MSH + 255) / 256, 256, 0, stream>>>(vb, vtb, MSH);

  // --- fused flash attention (no 1/sqrt(d) scaling, per reference) ---
  flash_attn_wmma<<<dim3(kS / 128, kNH, kB), 256, 0, stream>>>(
      qb, kpb, vtb, mask, attnb);

  // --- output projection + residual ---
  gemm_bf16_wmma<<<dim3(M / 128, kH / 128, 1), 256, 0, stream>>>(
      attnb, kH, 0, wot, 0, bo, src, 0, y1, nullptr, kH, 0, M, kH, kH, 0);

  // --- LN1 (fp32 + bf16 copy for the FFN GEMM) ---
  layernorm_k<<<M, 256, 0, stream>>>(y1, g1, be1, x1n, x1nb);

  // --- FFN ---
  gemm_bf16_wmma<<<dim3(M / 128, kPF / 128, 1), 256, 0, stream>>>(
      x1nb, kH, 0, w1t, 0, b1, nullptr, 0, nullptr, hb, kPF, 0, M, kPF, kH, 1);
  gemm_bf16_wmma<<<dim3(M / 128, kH / 128, 1), 256, 0, stream>>>(
      hb, kPF, 0, w2t, 0, b2, x1n, 0, y2, nullptr, kH, 0, M, kH, kPF, 0);

  // --- LN2 -> final fp32 output ---
  layernorm_k<<<M, 256, 0, stream>>>(y2, g2, be2, out, nullptr);
}
